// Decoder_32418413150393
// MI455X (gfx1250) — matrix operand order repaired and checked by reading, not compiled, not run on hardware
//
#include <hip/hip_runtime.h>

// ---------------------------------------------------------------------------
// ConvS2S decoder forward for MI455X (gfx1250, wave32, WMMA).
//  * All GEMMs: f16 WMMA (v_wmma_f32_16x16x32_f16), f32 accumulate.
//  * A and B tiles staged to LDS via GLOBAL_LOAD_ASYNC_TO_LDS_B128 (ASYNCcnt).
//  * Weights pre-transposed to f16 Bt[N][K]; activations keep an f16 shadow.
//  * Causal K=3 conv folded into one GEMM via contiguous padded windows.
// ---------------------------------------------------------------------------

typedef __attribute__((ext_vector_type(16))) _Float16 v16h;
typedef __attribute__((ext_vector_type(8)))  _Float16 v8h;
typedef __attribute__((ext_vector_type(8)))  float    v8f;

#define BM 128
#define BN 128
#define BK 32
#define LDP 40   // padded LDS row pitch in halves (80B) to spread banks

__device__ __forceinline__ v16h frag_from2(v8h lo, v8h hi) {
  v16h r;
#pragma unroll
  for (int j = 0; j < 8; ++j) { r[j] = lo[j]; r[j + 8] = hi[j]; }
  return r;
}

// Async DMA: 16B per lane, global -> LDS, tracked by ASYNCcnt.
__device__ __forceinline__ void async_b128(unsigned lds_off, const _Float16* g) {
  asm volatile("global_load_async_to_lds_b128 %0, %1, off"
               :: "v"(lds_off), "v"((unsigned long long)(size_t)g)
               : "memory");
}
__device__ __forceinline__ void wait_async0() {
  asm volatile("s_wait_asynccnt 0x0" ::: "memory");
}

// C[m,n] = (sum_k A16[m,k]*Bt[n,k] + bias[n] + other[m,n]) * alpha
// A16/Bt f16; optional f16 copy of C into C16. Row remap for conv im2col:
// rowIdx = (m>>tlog)*padT + (m & ((1<<tlog)-1)).
__global__ __launch_bounds__(256)
void k_gemm_wmma(const _Float16* __restrict__ A16, const _Float16* __restrict__ Bt,
                 const float* __restrict__ bias, const float* __restrict__ other,
                 float* __restrict__ C, _Float16* __restrict__ C16,
                 int M, int N, int Kd, int lda,
                 long long sA, long long sB, long long sC,
                 int tlog, int padT, float alpha) {
  __shared__ _Float16 Asm[2][BM][LDP];
  __shared__ _Float16 Bsm[2][BN][LDP];

  const int tid   = threadIdx.x;
  const int lane  = tid & 31;
  const int wid   = tid >> 5;
  const int waveM = wid & 3;   // 4 waves along M
  const int waveN = wid >> 2;  // 2 waves along N
  const int lr    = lane & 15;
  const int kh    = lane >> 4;

  const int m0 = blockIdx.y * BM;
  const int n0 = blockIdx.x * BN;
  const long long zA = (long long)blockIdx.z * sA;
  const long long zB = (long long)blockIdx.z * sB;
  const long long zC = (long long)blockIdx.z * sC;
  const unsigned tmask = (1u << tlog) - 1u;

  v8f acc[2][4];
#pragma unroll
  for (int i = 0; i < 2; ++i)
#pragma unroll
    for (int j = 0; j < 4; ++j)
#pragma unroll
      for (int v = 0; v < 8; ++v) acc[i][j][v] = 0.f;

  const int nK = Kd / BK;

  // Per-thread DMA descriptors: 2 A chunks + 2 B chunks of 16B (8 halves).
  const _Float16* abase[2];
  const _Float16* bbase[2];
  unsigned aoff[2][2], boff[2][2];   // [chunk][buf]
#pragma unroll
  for (int i = 0; i < 2; ++i) {
    int idx = tid * 2 + i;           // 0..511
    int row = idx >> 2;              // 0..127
    int c8  = idx & 3;               // chunk-of-8-halves in the 32-wide tile
    int m   = m0 + row;
    long long ri = (long long)(m >> tlog) * padT + (long long)(m & tmask);
    abase[i] = A16 + zA + ri * (long long)lda + c8 * 8;
    aoff[i][0] = (unsigned)(unsigned long long)(size_t)&Asm[0][row][c8 * 8];
    aoff[i][1] = (unsigned)(unsigned long long)(size_t)&Asm[1][row][c8 * 8];
    int gn = n0 + row;
    if (gn > N - 1) gn = N - 1;      // clamp: cols >= N are never stored
    bbase[i] = Bt + zB + (long long)gn * Kd + c8 * 8;
    boff[i][0] = (unsigned)(unsigned long long)(size_t)&Bsm[0][row][c8 * 8];
    boff[i][1] = (unsigned)(unsigned long long)(size_t)&Bsm[1][row][c8 * 8];
  }

  auto stage = [&](int kt, int buf) {
    const int k0 = kt * BK;
#pragma unroll
    for (int i = 0; i < 2; ++i) async_b128(aoff[i][buf], abase[i] + k0);
#pragma unroll
    for (int i = 0; i < 2; ++i) async_b128(boff[i][buf], bbase[i] + k0);
  };

  stage(0, 0);
  wait_async0();

  for (int kt = 0; kt < nK; ++kt) {
    __syncthreads();                 // publish LDS[cur] to all waves
    const int cur = kt & 1;
    if (kt + 1 < nK) stage(kt + 1, 1 - cur);   // DMA next tile during compute

    v16h af[2], bf[4];
#pragma unroll
    for (int i = 0; i < 2; ++i) {
      const _Float16* bp = &Asm[cur][waveM * 32 + i * 16 + lr][0];
      v8h lo = *(const v8h*)(bp + kh * 8);        // K = kh*8 .. +7
      v8h hi = *(const v8h*)(bp + 16 + kh * 8);   // K = 16+kh*8 .. +7
      af[i] = frag_from2(lo, hi);
    }
#pragma unroll
    for (int j = 0; j < 4; ++j) {
      const _Float16* bp = &Bsm[cur][waveN * 64 + j * 16 + lr][0];
      v8h lo = *(const v8h*)(bp + kh * 8);        // K = kh*8 .. +7
      v8h hi = *(const v8h*)(bp + 16 + kh * 8);   // K = 16+kh*8 .. +7
      bf[j] = frag_from2(lo, hi);
    }
#pragma unroll
    for (int i = 0; i < 2; ++i)
#pragma unroll
      for (int j = 0; j < 4; ++j)
        acc[i][j] = __builtin_amdgcn_wmma_f32_16x16x32_f16(
            false, af[i], false, bf[j], (short)0, acc[i][j], false, false);

    if (kt + 1 < nK) wait_async0();  // DMA done before next publish barrier
  }

  const int mB = m0 + waveM * 32;
  const int nB = n0 + waveN * 64;
#pragma unroll
  for (int i = 0; i < 2; ++i) {
#pragma unroll
    for (int j = 0; j < 4; ++j) {
      int col = nB + j * 16 + lr;
      if (col < N) {
        float bv = bias ? bias[col] : 0.f;
#pragma unroll
        for (int v = 0; v < 8; ++v) {
          int row = mB + i * 16 + kh * 8 + v;
          float val = acc[i][j][v] + bv;
          if (other) val += other[(long long)row * N + col];
          val *= alpha;
          C[zC + (long long)row * N + col] = val;
          if (C16) C16[zC + (long long)row * N + col] = (_Float16)val;
        }
      }
    }
  }
}

// ---------------------------- small kernels --------------------------------

__global__ void k_cvt(const float* __restrict__ in, _Float16* __restrict__ out,
                      long long n) {
  long long i = (long long)blockIdx.x * blockDim.x + threadIdx.x;
  if (i < n) out[i] = (_Float16)in[i];
}

// in: [z][Kr][Nc] f32 -> out: [z][Nc][Kr] f16 (transpose+convert)
__global__ void k_cvt_t(const float* __restrict__ in, _Float16* __restrict__ out,
                        int Kr, int Nc, long long sIn, long long sOut) {
  int n = blockIdx.x * blockDim.x + threadIdx.x;
  int k = blockIdx.y;
  long long zi = (long long)blockIdx.z * sIn;
  long long zo = (long long)blockIdx.z * sOut;
  if (n < Nc)
    out[zo + (long long)n * Kr + k] = (_Float16)in[zi + (long long)k * Nc + n];
}

// conv_w [L,2H=2048,H=1024,K=3] f32 -> [L][oc][k*H+ic] f16
__global__ void k_cvt_convw(const float* __restrict__ in,
                            _Float16* __restrict__ out, long long total) {
  long long i = (long long)blockIdx.x * blockDim.x + threadIdx.x;
  if (i >= total) return;
  int k = (int)(i % 3);
  long long t = i / 3;
  int ic = (int)(t & 1023); t >>= 10;
  int oc = (int)(t & 2047); t >>= 11;
  int l  = (int)t;
  out[((long long)l * 2048 + oc) * 3072 + (long long)k * 1024 + ic] =
      (_Float16)in[i];
}

// embedded[b*T+t][e] = tok_emb[tgt[b,t]][e] + pos_emb[t][e]   (E=512, T=512)
__global__ void k_embed(const int* __restrict__ tgt, const float* __restrict__ tok,
                        const float* __restrict__ pos, float* __restrict__ out,
                        _Float16* __restrict__ out16, long long n) {
  long long i = (long long)blockIdx.x * blockDim.x + threadIdx.x;
  if (i >= n) return;
  int e = (int)(i & 511);
  long long bt = i >> 9;
  int t = (int)(bt & 511);
  int id = tgt[bt];
  float v = tok[(long long)id * 512 + e] + pos[(long long)t * 512 + e];
  out[i] = v;
  out16[i] = (_Float16)v;
}

// padded16[b][t][h] (t in 0..513): t<2 -> PAD_VAL=1.0, else conv_input[b][t-2][h]
__global__ void k_pad(const float* __restrict__ ci, _Float16* __restrict__ out) {
  int x = blockIdx.x * blockDim.x + threadIdx.x;   // < 514*1024
  int b = blockIdx.y;
  if (x >= 514 * 1024) return;
  int h = x & 1023;
  int t = x >> 10;
  float v = (t < 2) ? 1.0f : ci[((long long)b * 512 + (t - 2)) * 1024 + h];
  out[(long long)b * 514 * 1024 + x] = (_Float16)v;
}

// GLU on [m][2H]: out[m][j] = x[m][j] * sigmoid(x[m][H+j])   (H=1024)
__global__ void k_glu(const float* __restrict__ x, float* __restrict__ out,
                      _Float16* __restrict__ out16, long long n) {
  long long i = (long long)blockIdx.x * blockDim.x + threadIdx.x;
  if (i >= n) return;
  int j = (int)(i & 1023);
  long long m = i >> 10;
  float a = x[m * 2048 + j];
  float g = x[m * 2048 + 1024 + j];
  float v = a * (1.f / (1.f + __expf(-g)));
  out[i] = v;
  out16[i] = (_Float16)v;
}

// ci = (ci + c2) * sqrt(0.5); also f16 shadow
__global__ void k_addscale(float* __restrict__ ci, _Float16* __restrict__ ci16,
                           const float* __restrict__ c2, long long n) {
  long long i = (long long)blockIdx.x * blockDim.x + threadIdx.x;
  if (i < n) {
    float v = (ci[i] + c2[i]) * 0.70710678118654752f;
    ci[i] = v;
    ci16[i] = (_Float16)v;
  }
}

// softmax over S=512 per row; writes f32 (tuple output) and f16 shadow
__global__ __launch_bounds__(256)
void k_softmax(const float* __restrict__ e, float* __restrict__ o,
               _Float16* __restrict__ o16) {
  __shared__ float red[256];
  long long row = blockIdx.x;
  const float* ep = e + row * 512;
  float* op = o + row * 512;
  _Float16* op16 = o16 + row * 512;
  int tid = threadIdx.x;
  float mx = -1e30f;
  for (int i = tid; i < 512; i += 256) mx = fmaxf(mx, ep[i]);
  red[tid] = mx; __syncthreads();
  for (int s = 128; s > 0; s >>= 1) {
    if (tid < s) red[tid] = fmaxf(red[tid], red[tid + s]);
    __syncthreads();
  }
  mx = red[0]; __syncthreads();
  float sum = 0.f;
  for (int i = tid; i < 512; i += 256) {
    float v = __expf(ep[i] - mx);
    op[i] = v;
    sum += v;
  }
  red[tid] = sum; __syncthreads();
  for (int s = 128; s > 0; s >>= 1) {
    if (tid < s) red[tid] += red[tid + s];
    __syncthreads();
  }
  float inv = 1.f / red[0];
  for (int i = tid; i < 512; i += 256) {
    float v = op[i] * inv;
    op[i] = v;
    op16[i] = (_Float16)v;
  }
}

// ------------------------------- host --------------------------------------

static inline void gemm(hipStream_t st, const _Float16* A16, const _Float16* Bt,
                        const float* bias, const float* other, float* C,
                        _Float16* C16, int M, int N, int Kd, int lda,
                        long long sA, long long sB, long long sC, int batches,
                        int tlog, int padT, float alpha) {
  dim3 g((N + BN - 1) / BN, M / BM, batches);
  k_gemm_wmma<<<g, 256, 0, st>>>(A16, Bt, bias, other, C, C16, M, N, Kd, lda,
                                 sA, sB, sC, tlog, padT, alpha);
}

extern "C" void kernel_launch(void* const* d_in, const int* in_sizes, int n_in,
                              void* d_out, int out_size, void* d_ws, size_t ws_size,
                              hipStream_t stream) {
  const int*   tgt          = (const int*)  d_in[0];
  const float* enc_conved   = (const float*)d_in[1];
  const float* enc_combined = (const float*)d_in[2];
  const float* tok_emb      = (const float*)d_in[3];
  const float* pos_emb      = (const float*)d_in[4];
  const float* emb2hid_w    = (const float*)d_in[5];
  const float* emb2hid_b    = (const float*)d_in[6];
  const float* hid2emb_w    = (const float*)d_in[7];
  const float* hid2emb_b    = (const float*)d_in[8];
  const float* ah2e_w       = (const float*)d_in[9];
  const float* ah2e_b       = (const float*)d_in[10];
  const float* ae2h_w       = (const float*)d_in[11];
  const float* ae2h_b       = (const float*)d_in[12];
  const float* fc_w         = (const float*)d_in[13];
  const float* fc_b         = (const float*)d_in[14];
  const float* conv_w       = (const float*)d_in[15];
  const float* conv_b       = (const float*)d_in[16];

  constexpr int Bz = 16, T = 512, S = 512, V = 10000, E = 512, H = 1024, Ln = 6;
  constexpr long long BT = (long long)Bz * T;
  const float scale = 0.70710678118654752f;

  char* wp = (char*)d_ws;
  auto alloc = [&](size_t bytes) -> void* {
    void* p = (void*)wp;
    wp += (bytes + 255) & ~(size_t)255;
    return p;
  };

  // f16 transposed weights Bt[N][K]
  _Float16* wt_e2h  = (_Float16*)alloc((size_t)H * E * 2);   // [H][E]
  _Float16* wt_h2e  = (_Float16*)alloc((size_t)E * H * 2);   // [E][H]
  _Float16* wt_ah2e = (_Float16*)alloc((size_t)E * H * 2);   // [E][H]
  _Float16* wt_ae2h = (_Float16*)alloc((size_t)H * E * 2);   // [H][E]
  _Float16* wt_fc   = (_Float16*)alloc((size_t)V * E * 2);   // [V][E]
  _Float16* wt_conv = (_Float16*)alloc((size_t)Ln * 2 * H * 3 * H * 2);
  _Float16* encC16  = (_Float16*)alloc((size_t)Bz * S * E * 2);  // [B][S][E]
  _Float16* encCT16 = (_Float16*)alloc((size_t)Bz * E * S * 2);  // [B][E][S]
  // f32 activations + f16 shadows (GEMM A operands)
  float*    embedded   = (float*)   alloc((size_t)BT * E * 4);
  _Float16* embedded16 = (_Float16*)alloc((size_t)BT * E * 2);
  float*    conv_in    = (float*)   alloc((size_t)BT * H * 4);
  _Float16* convin16   = (_Float16*)alloc((size_t)BT * H * 2);
  _Float16* padded16   = (_Float16*)alloc((size_t)Bz * (T + 2) * H * 2);
  float*    conv_out   = (float*)   alloc((size_t)BT * 2 * H * 4);
  float*    glu        = (float*)   alloc((size_t)BT * H * 4);
  _Float16* glu16      = (_Float16*)alloc((size_t)BT * H * 2);
  float*    combined   = (float*)   alloc((size_t)BT * E * 4);
  _Float16* combined16 = (_Float16*)alloc((size_t)BT * E * 2);
  float*    energy     = (float*)   alloc((size_t)Bz * T * S * 4);
  _Float16* attn16     = (_Float16*)alloc((size_t)Bz * T * S * 2);
  float*    attended   = (float*)   alloc((size_t)BT * E * 4);
  _Float16* attended16 = (_Float16*)alloc((size_t)BT * E * 2);
  float*    conved2    = (float*)   alloc((size_t)BT * H * 4);
  float*    convedF    = (float*)   alloc((size_t)BT * E * 4);
  _Float16* convedF16  = (_Float16*)alloc((size_t)BT * E * 2);

  float* out  = (float*)d_out;
  float* attn = out + (long long)BT * V;   // tuple output #2: [B,T,S] f32

  // ---- weight / encoder conversions ----
  k_cvt_t<<<dim3((H + 255) / 256, E, 1), 256, 0, stream>>>(emb2hid_w, wt_e2h, E, H, 0, 0);
  k_cvt_t<<<dim3((E + 255) / 256, H, 1), 256, 0, stream>>>(hid2emb_w, wt_h2e, H, E, 0, 0);
  k_cvt_t<<<dim3((E + 255) / 256, H, 1), 256, 0, stream>>>(ah2e_w,    wt_ah2e, H, E, 0, 0);
  k_cvt_t<<<dim3((H + 255) / 256, E, 1), 256, 0, stream>>>(ae2h_w,    wt_ae2h, E, H, 0, 0);
  k_cvt_t<<<dim3((V + 255) / 256, E, 1), 256, 0, stream>>>(fc_w,      wt_fc,   E, V, 0, 0);
  k_cvt_t<<<dim3((E + 255) / 256, S, Bz), 256, 0, stream>>>(
      enc_combined, encCT16, S, E, (long long)S * E, (long long)E * S);
  { long long n = (long long)Bz * S * E;
    k_cvt<<<(unsigned)((n + 255) / 256), 256, 0, stream>>>(enc_conved, encC16, n); }
  { long long n = (long long)Ln * 2 * H * H * 3;
    k_cvt_convw<<<(unsigned)((n + 255) / 256), 256, 0, stream>>>(conv_w, wt_conv, n); }
  { long long n = BT * E;
    k_embed<<<(unsigned)((n + 255) / 256), 256, 0, stream>>>(
        tgt, tok_emb, pos_emb, embedded, embedded16, n); }

  // conv_input = embedded @ emb2hid + b   [BT,H]
  gemm(stream, embedded16, wt_e2h, emb2hid_b, nullptr, conv_in, nullptr,
       (int)BT, H, E, E, 0, 0, 0, 1, 30, 0, 1.f);

  for (int l = 0; l < Ln; ++l) {
    k_pad<<<dim3(((T + 2) * H + 255) / 256, Bz), 256, 0, stream>>>(conv_in, padded16);

    // conv as GEMM: M=BT, N=2H, K=3H; A row = contiguous 3H window in padded16
    gemm(stream, padded16, wt_conv + (size_t)l * 2 * H * 3 * H, conv_b + l * 2 * H,
         nullptr, conv_out, nullptr, (int)BT, 2 * H, 3 * H, H, 0, 0, 0, 1,
         /*tlog=*/9, /*padT=*/T + 2, 1.f);

    { long long n = BT * H;
      k_glu<<<(unsigned)((n + 255) / 256), 256, 0, stream>>>(conv_out, glu, glu16, n); }

    // combined = (glu @ attn_hid2emb + b + embedded) * scale   [BT,E]
    gemm(stream, glu16, wt_ah2e, ah2e_b, embedded, combined, combined16,
         (int)BT, E, H, H, 0, 0, 0, 1, 30, 0, scale);

    // energy[b] = combined[b] @ enc_conved[b]^T   [T,S]
    gemm(stream, combined16, encC16, nullptr, nullptr, energy, nullptr,
         T, S, E, E, (long long)T * E, (long long)S * E, (long long)T * S, Bz,
         30, 0, 1.f);

    k_softmax<<<(unsigned)BT, 256, 0, stream>>>(energy, attn, attn16);

    // attended[b] = attn[b] @ enc_combined[b]   [T,E]
    gemm(stream, attn16, encCT16, nullptr, nullptr, attended, attended16,
         T, E, S, S, (long long)T * S, (long long)E * S, (long long)T * E, Bz,
         30, 0, 1.f);

    // conved2 = (attended @ attn_emb2hid + b + glu) * scale   [BT,H]
    gemm(stream, attended16, wt_ae2h, ae2h_b, glu, conved2, nullptr,
         (int)BT, H, E, E, 0, 0, 0, 1, 30, 0, scale);

    // conv_input = (conved2 + conv_input) * scale (+ f16 shadow)
    { long long n = BT * H;
      k_addscale<<<(unsigned)((n + 255) / 256), 256, 0, stream>>>(
          conv_in, convin16, conved2, n); }
  }

  // convedF = conv_input @ hid2emb + b   [BT,E]
  gemm(stream, convin16, wt_h2e, hid2emb_b, nullptr, convedF, convedF16,
       (int)BT, E, H, H, 0, 0, 0, 1, 30, 0, 1.f);

  // output = convedF @ fc_out + b   [BT,V] (N=10000 -> guarded tiles)
  gemm(stream, convedF16, wt_fc, fc_b, nullptr, out, nullptr,
       (int)BT, V, E, E, 0, 0, 0, 1, 30, 0, 1.f);
}
